// VoxelResBackBone8xVoxelNeXt2D_27401891348896
// MI455X (gfx1250) — matrix operand order repaired and checked by reading, not compiled, not run on hardware
//
#include <hip/hip_runtime.h>
#include <stdint.h>

// ---------------------------------------------------------------------------
// CDNA5 (gfx1250) implicit-GEMM conv backbone using v_wmma_f32_16x16x32_f16.
// wave32: one wave computes a (16*PT pixels) x (32 couts) macro-tile as a
// PT x 2 grid of 16x16 WMMA tiles, K-loop over 9 taps x (Cin/32) chunks.
// fp16 activations / weights, fp32 accumulation. 2x2 register blocking gives
// 4 independent WMMAs per 4KB of fragment loads (16 FLOP/B from L2).
// ---------------------------------------------------------------------------

typedef __attribute__((ext_vector_type(16))) _Float16 v16h;
typedef __attribute__((ext_vector_type(8)))  _Float16 h8;
typedef __attribute__((ext_vector_type(8)))  float    v8f;

#define GH 512
#define GW 512
#define GC 32
#define GNVOX 60000

// ---------------- utility kernels ----------------

__global__ void fill0_u32(uint32_t* __restrict__ p, size_t n) {
  size_t i = (size_t)blockIdx.x * blockDim.x + threadIdx.x;
  if (i < n) p[i] = 0u;
}

__global__ void scatter_kernel(const float* __restrict__ feats,
                               const int* __restrict__ coords,
                               _Float16* __restrict__ x, float* __restrict__ m,
                               int nvox) {
  int i = blockIdx.x * blockDim.x + threadIdx.x;
  if (i >= nvox) return;
  int y  = coords[i * 3 + 1];
  int xx = coords[i * 3 + 2];
  size_t pix = (size_t)y * GW + xx;
  m[pix] = 1.0f;
  const float* f = feats + (size_t)i * GC;
  _Float16* dst  = x + pix * GC;
  for (int c = 0; c < GC; ++c) dst[c] = (_Float16)f[c];
}

// Re-pack HWIO fp32 weights into WMMA B-fragment order (f16).
// dst index i = ((tile*32 + lane)*16 + h)
//   tile = ((coTile*9 + kpos)*ciChunks + cc)
//   K = (h&8)*2 + 8*(lane>>4) + (h&7) ; ci = cc*32 + K ; co = coTile*16 + (lane&15)
__global__ void pack_w(const float* __restrict__ w, _Float16* __restrict__ dst,
                       int Cin, int Cout) {
  size_t total = (size_t)9 * Cin * Cout;
  size_t i = (size_t)blockIdx.x * blockDim.x + threadIdx.x;
  if (i >= total) return;
  int h    = (int)(i & 15);
  int lane = (int)((i >> 4) & 31);
  size_t tile = i >> 9;
  int ciChunks = Cin >> 5;
  int cc     = (int)(tile % ciChunks);
  size_t t2  = tile / ciChunks;
  int kpos   = (int)(t2 % 9);
  int coTile = (int)(t2 / 9);
  int K  = ((h & 8) << 1) + ((lane >> 4) << 3) + (h & 7);
  int ci = (cc << 5) + K;
  int co = (coTile << 4) + (lane & 15);
  dst[i] = (_Float16)w[((size_t)kpos * Cin + ci) * Cout + co];
}

__global__ void dilate_mask(const float* __restrict__ m, float* __restrict__ nm,
                            int Hi, int Wi, int Ho, int Wo, int stride) {
  int i = blockIdx.x * blockDim.x + threadIdx.x;
  if (i >= Ho * Wo) return;
  int yo = i / Wo, xo = i - yo * Wo;
  float s = 0.0f;
  for (int dy = 0; dy < 3; ++dy)
    for (int dx = 0; dx < 3; ++dx) {
      int yi = yo * stride + dy - 1;
      int xi = xo * stride + dx - 1;
      if (yi >= 0 && yi < Hi && xi >= 0 && xi < Wi && m[(size_t)yi * Wi + xi] > 0.0f)
        s = 1.0f;
    }
  nm[i] = s;
}

// comb = x4 + zero-stuffed-up(x5, 2) + zero-stuffed-up(x6, 4)  at 64x64x256
__global__ void combine_feat(const _Float16* __restrict__ x4,
                             const _Float16* __restrict__ x5,
                             const _Float16* __restrict__ x6,
                             _Float16* __restrict__ comb) {
  int i = blockIdx.x * blockDim.x + threadIdx.x;
  if (i >= 64 * 64 * 256) return;
  int c = i & 255;
  int x = (i >> 8) & 63;
  int y = i >> 14;
  float v = (float)x4[i];
  if (((y | x) & 1) == 0) v += (float)x5[(((size_t)(y >> 1) * 32) + (x >> 1)) * 256 + c];
  if (((y | x) & 3) == 0) v += (float)x6[(((size_t)(y >> 2) * 16) + (x >> 2)) * 256 + c];
  comb[i] = (_Float16)v;
}

__global__ void combine_mask(const float* __restrict__ m4,
                             const float* __restrict__ m5,
                             const float* __restrict__ m6,
                             float* __restrict__ mc) {
  int i = blockIdx.x * blockDim.x + threadIdx.x;
  if (i >= 64 * 64) return;
  int x = i & 63, y = i >> 6;
  float v = m4[i];
  if (((y | x) & 1) == 0) v = fmaxf(v, m5[(y >> 1) * 32 + (x >> 1)]);
  if (((y | x) & 3) == 0) v = fmaxf(v, m6[(y >> 2) * 16 + (x >> 2)]);
  mc[i] = v;
}

// ---------------- fused conv3x3 + bias + BN + mask + residual + ReLU ----------------
// One wave per (16*PT W-pixels, 1 H-row, 32 couts) macro-tile; PT x 2 WMMA tiles.
// A fragment (16x32 f16): lane L handles pixel m=L&15, khalf=L>>4;
//   halves 0..7  -> K = 8*khalf + 0..7   (contiguous 16B load)
//   halves 8..15 -> K = 16 + 8*khalf + 0..7 (contiguous 16B load)
// B fragment: packed by pack_w so each lane reads 16 contiguous halves (2x16B).
template <int PT>
__global__ __launch_bounds__(32)
void conv3x3_wmma(const _Float16* __restrict__ in, const _Float16* __restrict__ wpk,
                  const float* __restrict__ bias, const float* __restrict__ scale,
                  const float* __restrict__ shift, const float* __restrict__ mask,
                  const _Float16* __restrict__ residual,
                  _Float16* __restrict__ outh, float* __restrict__ outf,
                  int H, int W, int Cin, int Ho, int Wo, int Cout, int stride) {
  const int lane  = threadIdx.x;
  const int mpix  = lane & 15;
  const int khalf = lane >> 4;
  const int xo0 = blockIdx.x * (16 * PT);
  const int yo  = blockIdx.y;
  const int co0 = blockIdx.z << 5;           // 32 couts per wave
  const int ciChunks = Cin >> 5;

  v8f acc[PT][2];
#pragma unroll
  for (int pt = 0; pt < PT; ++pt) {
    acc[pt][0] = (v8f){};
    acc[pt][1] = (v8f){};
  }
  const h8 zero8 = {};

  for (int kpos = 0; kpos < 9; ++kpos) {
    const int ky = kpos / 3;
    const int kx = kpos - ky * 3;
    const int yi = yo * stride + ky - 1;
    const bool inbY = (yi >= 0) & (yi < H);

    const _Float16* src[PT];
    bool inb[PT];
#pragma unroll
    for (int pt = 0; pt < PT; ++pt) {
      const int xi = (xo0 + pt * 16 + mpix) * stride + kx - 1;
      inb[pt] = inbY & (xi >= 0) & (xi < W);
      const int yc = inb[pt] ? yi : 0;
      const int xc = inb[pt] ? xi : 0;
      src[pt] = in + ((size_t)yc * W + xc) * Cin;
    }

    // coTile pair for this wave
    const size_t tb0 = ((size_t)((co0 >> 4) + 0) * 9 + kpos) * ciChunks;
    const size_t tb1 = ((size_t)((co0 >> 4) + 1) * 9 + kpos) * ciChunks;

    for (int cc = 0; cc < ciChunks; ++cc) {
      const int ci0 = cc << 5;
      v16h a[PT];
#pragma unroll
      for (int pt = 0; pt < PT; ++pt) {
        h8 a0 = *(const h8*)(src[pt] + ci0 + (khalf << 3));
        h8 a1 = *(const h8*)(src[pt] + ci0 + 16 + (khalf << 3));
        if (!inb[pt]) { a0 = zero8; a1 = zero8; }
        ((h8*)&a[pt])[0] = a0;
        ((h8*)&a[pt])[1] = a1;
      }
      v16h b[2];
      {
        const _Float16* bp0 = wpk + (((tb0 + cc) << 5) + lane) * 16;
        const _Float16* bp1 = wpk + (((tb1 + cc) << 5) + lane) * 16;
        ((h8*)&b[0])[0] = *(const h8*)(bp0);
        ((h8*)&b[0])[1] = *(const h8*)(bp0 + 8);
        ((h8*)&b[1])[0] = *(const h8*)(bp1);
        ((h8*)&b[1])[1] = *(const h8*)(bp1 + 8);
      }
#pragma unroll
      for (int pt = 0; pt < PT; ++pt) {
#pragma unroll
        for (int ct = 0; ct < 2; ++ct) {
          acc[pt][ct] = __builtin_amdgcn_wmma_f32_16x16x32_f16(
              false, a[pt], false, b[ct], (short)0, acc[pt][ct], false, false);
        }
      }
    }
  }

  // C/D layout: reg r, lanes 0-15 -> M=r, lanes 16-31 -> M=8+r; N = lane&15
  const int n = lane & 15;
#pragma unroll
  for (int ct = 0; ct < 2; ++ct) {
    const int co = co0 + ct * 16 + n;
    const float sc = scale[co];
    const float sh = shift[co];
    const float bs = bias ? bias[co] : 0.0f;
#pragma unroll
    for (int pt = 0; pt < PT; ++pt) {
#pragma unroll
      for (int r = 0; r < 8; ++r) {
        const int mm = pt * 16 + r + (khalf << 3);
        const size_t pix = (size_t)yo * Wo + (xo0 + mm);
        float v = acc[pt][ct][r] + bs;
        v = v * sc + sh;                       // BN affine
        if (mask) v *= mask[pix];              // mask
        if (residual) v += (float)residual[pix * Cout + co];
        v = fmaxf(v, 0.0f);                    // ReLU (all convs in this net)
        if (outf) outf[pix * Cout + co] = v;
        else      outh[pix * Cout + co] = (_Float16)v;
      }
    }
  }
}

// ---------------- host-side orchestration ----------------

struct BlkP  { const float *b1, *b2, *s1, *h1, *s2, *h2, *w1, *w2; };
struct DownP { const float *s, *h, *w; };

static void launch_pack(hipStream_t st, const float* w, _Float16* dst, int Cin, int Cout) {
  size_t total = (size_t)9 * Cin * Cout;
  pack_w<<<dim3((unsigned)((total + 255) / 256)), dim3(256), 0, st>>>(w, dst, Cin, Cout);
}

static void launch_conv(hipStream_t st, const _Float16* in, const _Float16* wpk,
                        const float* bias, const float* sc, const float* sh,
                        const float* mask, const _Float16* res,
                        _Float16* outh, float* outf,
                        int H, int W, int Cin, int Cout, int stride) {
  int Ho = H / stride, Wo = W / stride;
  if ((Wo & 31) == 0) {
    dim3 g((unsigned)(Wo / 32), (unsigned)Ho, (unsigned)(Cout / 32));
    conv3x3_wmma<2><<<g, dim3(32), 0, st>>>(in, wpk, bias, sc, sh, mask, res, outh, outf,
                                            H, W, Cin, Ho, Wo, Cout, stride);
  } else {
    dim3 g((unsigned)(Wo / 16), (unsigned)Ho, (unsigned)(Cout / 32));
    conv3x3_wmma<1><<<g, dim3(32), 0, st>>>(in, wpk, bias, sc, sh, mask, res, outh, outf,
                                            H, W, Cin, Ho, Wo, Cout, stride);
  }
}

extern "C" void kernel_launch(void* const* d_in, const int* in_sizes, int n_in,
                              void* d_out, int out_size, void* d_ws, size_t ws_size,
                              hipStream_t stream) {
  (void)in_sizes; (void)n_in; (void)out_size; (void)ws_size;

  // ---- input leaves (jax sorted-key flatten order) ----
  int idx = 0;
  const float* feats = (const float*)d_in[idx++];
  const int*   coords = (const int*)d_in[idx++];
  auto takeF   = [&]() { return (const float*)d_in[idx++]; };
  auto takeBlk = [&]() {
    BlkP b;
    b.b1 = takeF(); b.b2 = takeF();
    b.s1 = takeF(); b.h1 = takeF();
    b.s2 = takeF(); b.h2 = takeF();
    b.w1 = takeF(); b.w2 = takeF();
    return b;
  };
  auto takeDown = [&]() { DownP d; d.s = takeF(); d.h = takeF(); d.w = takeF(); return d; };

  BlkP c1[3];  for (int i = 0; i < 3; ++i) c1[i] = takeBlk();
  BlkP c2b[4]; for (int i = 0; i < 4; ++i) c2b[i] = takeBlk();
  DownP c2d = takeDown();
  BlkP c3b[6]; for (int i = 0; i < 6; ++i) c3b[i] = takeBlk();
  DownP c3d = takeDown();
  BlkP c4b[3]; for (int i = 0; i < 3; ++i) c4b[i] = takeBlk();
  DownP c4d = takeDown();
  BlkP c5b[3]; for (int i = 0; i < 3; ++i) c5b[i] = takeBlk();
  DownP c5d = takeDown();
  BlkP c6b[3]; for (int i = 0; i < 3; ++i) c6b[i] = takeBlk();
  DownP c6d = takeDown();
  const float *co_s = takeF(), *co_h = takeF(), *co_w = takeF();
  const float *sh_b = takeF(), *sh_s = takeF(), *sh_h = takeF(), *sh_w = takeF();
  // batch_size leaf ignored (BATCH == 1)

  // ---- workspace bump allocator ----
  char* wsb = (char*)d_ws;
  size_t off = 0;
  auto alloc  = [&](size_t bytes) { off = (off + 255) & ~(size_t)255; void* p = wsb + off; off += bytes; return p; };
  auto allocH = [&](size_t e) { return (_Float16*)alloc(e * 2); };
  auto allocM = [&](size_t e) { return (float*)alloc(e * 4); };

  _Float16 *x512[3], *x256[3], *x128[3], *x64b[3], *x32b[3], *x16b[3];
  for (int i = 0; i < 3; ++i) x512[i] = allocH((size_t)512 * 512 * 32);
  for (int i = 0; i < 3; ++i) x256[i] = allocH((size_t)256 * 256 * 64);
  for (int i = 0; i < 3; ++i) x128[i] = allocH((size_t)128 * 128 * 128);
  for (int i = 0; i < 3; ++i) x64b[i] = allocH((size_t)64 * 64 * 256);
  for (int i = 0; i < 3; ++i) x32b[i] = allocH((size_t)32 * 32 * 256);
  for (int i = 0; i < 3; ++i) x16b[i] = allocH((size_t)16 * 16 * 256);
  float* m1 = allocM(512 * 512);
  float* m2 = allocM(256 * 256);
  float* m3 = allocM(128 * 128);
  float* m4 = allocM(64 * 64);
  float* m5 = allocM(32 * 32);
  float* m6 = allocM(16 * 16);
  float* mc = allocM(64 * 64);
  float* mo = allocM(64 * 64);
  _Float16* wbuf = allocH((size_t)9 * 256 * 256);  // largest packed weight

  // ---- scatter pillars into dense grid + mask ----
  {
    size_t nw = (size_t)512 * 512 * 32 * 2 / 4;  // u32 words of x512[0]
    fill0_u32<<<dim3((unsigned)((nw + 255) / 256)), dim3(256), 0, stream>>>((uint32_t*)x512[0], nw);
    size_t mw = (size_t)512 * 512;
    fill0_u32<<<dim3((unsigned)((mw + 255) / 256)), dim3(256), 0, stream>>>((uint32_t*)m1, mw);
    scatter_kernel<<<dim3((GNVOX + 255) / 256), dim3(256), 0, stream>>>(feats, coords, x512[0], m1, GNVOX);
  }

  // ---- residual stage runner ----
  auto runBlock = [&](const BlkP& p, _Float16*& cur, _Float16* tmp, _Float16*& spare,
                      const float* mask, int H, int C) {
    launch_pack(stream, p.w1, wbuf, C, C);
    launch_conv(stream, cur, wbuf, p.b1, p.s1, p.h1, mask, nullptr, tmp, nullptr, H, H, C, C, 1);
    launch_pack(stream, p.w2, wbuf, C, C);
    launch_conv(stream, tmp, wbuf, p.b2, p.s2, p.h2, mask, cur, spare, nullptr, H, H, C, C, 1);
    _Float16* t = cur; cur = spare; spare = t;
  };
  auto runStage = [&](const BlkP* blks, int nb, _Float16** bufs,
                      const float* mask, int H, int C) -> _Float16* {
    _Float16 *cur = bufs[0], *tmp = bufs[1], *spare = bufs[2];
    for (int i = 0; i < nb; ++i) runBlock(blks[i], cur, tmp, spare, mask, H, C);
    return cur;
  };
  auto runDown = [&](const DownP& d, const _Float16* in, _Float16* out,
                     const float* mIn, float* mOut, int Hi, int Cin, int Cout) {
    int Ho = Hi / 2;
    dilate_mask<<<dim3((Ho * Ho + 255) / 256), dim3(256), 0, stream>>>(mIn, mOut, Hi, Hi, Ho, Ho, 2);
    launch_pack(stream, d.w, wbuf, Cin, Cout);
    launch_conv(stream, in, wbuf, nullptr, d.s, d.h, mOut, nullptr, out, nullptr, Hi, Hi, Cin, Cout, 2);
  };

  // ---- backbone ----
  _Float16* x1 = runStage(c1, 3, x512, m1, 512, 32);

  runDown(c2d, x1, x256[0], m1, m2, 512, 32, 64);
  _Float16* x2 = runStage(c2b, 4, x256, m2, 256, 64);

  runDown(c3d, x2, x128[0], m2, m3, 256, 64, 128);
  _Float16* x3 = runStage(c3b, 6, x128, m3, 128, 128);

  runDown(c4d, x3, x64b[0], m3, m4, 128, 128, 256);
  _Float16* x4 = runStage(c4b, 3, x64b, m4, 64, 256);

  runDown(c5d, x4, x32b[0], m4, m5, 64, 256, 256);
  _Float16* x5 = runStage(c5b, 3, x32b, m5, 32, 256);

  runDown(c6d, x5, x16b[0], m5, m6, 32, 256, 256);
  _Float16* x6 = runStage(c6b, 3, x16b, m6, 16, 256);

  // ---- FPN combine at 64x64 ----
  _Float16* comb   = (x4 == x64b[0]) ? x64b[2] : x64b[0];
  _Float16* outTmp = x64b[1];
  combine_feat<<<dim3((64 * 64 * 256 + 255) / 256), dim3(256), 0, stream>>>(x4, x5, x6, comb);
  combine_mask<<<dim3((64 * 64 + 255) / 256), dim3(256), 0, stream>>>(m4, m5, m6, mc);
  dilate_mask<<<dim3((64 * 64 + 255) / 256), dim3(256), 0, stream>>>(mc, mo, 64, 64, 64, 64, 1);

  // conv_out: relu(bn(conv(comb)) * mo)
  launch_pack(stream, co_w, wbuf, 256, 256);
  launch_conv(stream, comb, wbuf, nullptr, co_s, co_h, mo, nullptr, outTmp, nullptr, 64, 64, 256, 256, 1);

  // shared: relu(bn(conv(out) + b) * mo) -> fp32 d_out
  launch_pack(stream, sh_w, wbuf, 256, 256);
  launch_conv(stream, outTmp, wbuf, sh_b, sh_s, sh_h, mo, nullptr, nullptr, (float*)d_out,
              64, 64, 256, 256, 1);
}
